// GCN_17781164606120
// MI455X (gfx1250) — matrix operand order repaired and checked by reading, not compiled, not run on hardware
//
#include <hip/hip_runtime.h>

typedef __attribute__((ext_vector_type(16))) _Float16 v16h;
typedef __attribute__((ext_vector_type(8)))  _Float16 v8h;
typedef __attribute__((ext_vector_type(4)))  _Float16 v4h;
typedef __attribute__((ext_vector_type(8)))  float    v8f;

#define K_DIM 256
#define N_DIM 128
#define M_TILE 80          // rows per block; 100000 = 1250 * 80 (exact)
#define M_SUB  5           // 5 sub-tiles of 16 rows

// ---------------------------------------------------------------------------
// Kernel 1: W [256x128] f32 (row-major, KxN) -> Wt [128x256] f16 (NxK)
// so WMMA B fragments become contiguous-in-K 16-half loads.
// ---------------------------------------------------------------------------
__global__ __launch_bounds__(256) void wt_convert_kernel(const float* __restrict__ W,
                                                         _Float16* __restrict__ Wt) {
    int idx = blockIdx.x * blockDim.x + threadIdx.x;
    if (idx >= K_DIM * N_DIM) return;
    int n = idx >> 8;        // 0..127
    int k = idx & 255;       // 0..255
    Wt[idx] = (_Float16)W[k * N_DIM + n];
}

// ---------------------------------------------------------------------------
// Kernel 2: h = x @ W via v_wmma_f32_16x16x32_f16.
// Block = 256 threads = 8 waves; block owns an 80-row M tile, wave w owns the
// 16-column N tile at 16*w. x strip staged f32->f16 into padded LDS once;
// each wave holds its 8 B fragments in registers across 5 M sub-tiles
// -> 40 WMMAs per wave, Wt fetched once per block.
// ---------------------------------------------------------------------------
__global__ __launch_bounds__(256) void gemm_kernel(const float* __restrict__ x,
                                                   const _Float16* __restrict__ Wt,
                                                   _Float16* __restrict__ h) {
    // 80 rows x 256 K in f16, +8 halfs row padding (row stride 528B = 132
    // dwords -> bank phase advances 4/row -> conflict-free column access).
    __shared__ _Float16 Ash[M_TILE][K_DIM + 8];

    const int tid  = threadIdx.x;
    const int lane = tid & 31;
    const int wave = tid >> 5;
    const int m0   = blockIdx.x * M_TILE;

    // Cooperative stage: 80*16 = 1280 chunks of 16 floats; 5 chunks/thread.
#pragma unroll
    for (int it = 0; it < M_SUB; ++it) {
        const int chunk = tid + it * 256;        // 0..1279
        const int lm = chunk >> 4;               // row in tile 0..79
        const int lk = (chunk & 15) << 4;        // 0,16,...,240
        const float* src = x + (size_t)(m0 + lm) * K_DIM + lk;
        v8h o0, o1;
#pragma unroll
        for (int j = 0; j < 8; ++j) o0[j] = (_Float16)src[j];
#pragma unroll
        for (int j = 0; j < 8; ++j) o1[j] = (_Float16)src[8 + j];
        *(v8h*)&Ash[lm][lk]     = o0;
        *(v8h*)&Ash[lm][lk + 8] = o1;
    }
    __syncthreads();

    const int am   = lane & 15;   // A row within 16-row sub-tile
    const int half = lane >> 4;   // lane half selects K sub-range
    const int n0   = wave << 4;
    const int bn   = n0 + (lane & 15);   // B column

    // B 32x16 f16 layout: lane half hh holds 16 contiguous K at kk+16*hh for
    // column N = lane%16 -> contiguous in Wt (NxK). Load all 8 frags once.
    v16h bfrag[8];
#pragma unroll
    for (int kt = 0; kt < 8; ++kt)
        bfrag[kt] = *(const v16h*)(Wt + (size_t)bn * K_DIM + kt * 32 + half * 16);

#pragma unroll
    for (int mt = 0; mt < M_SUB; ++mt) {
        v8f acc = {};
#pragma unroll
        for (int kt = 0; kt < 8; ++kt) {
            const int kk = kt * 32;
            // A 16x32 f16 layout (matches B): lane half hh holds
            // K = kk+16*hh..+15 contiguous for row M = lane%16.
            union { v16h v; v8h p[2]; } a;
            a.p[0] = *(const v8h*)&Ash[mt * 16 + am][kk + half * 16];
            a.p[1] = *(const v8h*)&Ash[mt * 16 + am][kk + half * 16 + 8];
            acc = __builtin_amdgcn_wmma_f32_16x16x32_f16(
                /*neg_a=*/false, a.v, /*neg_b=*/false, bfrag[kt],
                /*c_mod=*/(short)0, acc, /*reuse_a=*/false, /*reuse_b=*/false);
        }
        // C layout: VGPR r, lane L -> M = r + 8*(L/16), N = L%16. Store f16 h.
        const int crow = m0 + mt * 16 + (half << 3);
        const int ccol = n0 + (lane & 15);
#pragma unroll
        for (int r = 0; r < 8; ++r)
            h[(size_t)(crow + r) * N_DIM + ccol] = (_Float16)acc[r];
    }
}

// ---------------------------------------------------------------------------
// Kernel 3: SpMM out[row] += val * h[col]. Wave-per-edge; lane owns 4 features
// (32 lanes * 4 = 128). Gather = coalesced 256B f16 line (L2-resident since
// h is 25.6MB << 192MB L2), scatter = global_atomic_add_f32 into zeroed out.
// ---------------------------------------------------------------------------
__global__ __launch_bounds__(256) void spmm_kernel(const float* __restrict__ vals,
                                                   const int* __restrict__ rows,
                                                   const int* __restrict__ cols,
                                                   const _Float16* __restrict__ h,
                                                   float* __restrict__ out,
                                                   int nEdges) {
    long long gid = (long long)blockIdx.x * blockDim.x + threadIdx.x;
    int e = (int)(gid >> 5);
    if (e >= nEdges) return;
    int lane = (int)(gid & 31);

    int   r = rows[e];
    int   c = cols[e];
    float v = vals[e];

    v4h hv = *(const v4h*)(h + ((size_t)c << 7) + (lane << 2));   // 8B b64 load
    float* op = out + ((size_t)r << 7) + (lane << 2);
    atomicAdd(op + 0, v * (float)hv[0]);
    atomicAdd(op + 1, v * (float)hv[1]);
    atomicAdd(op + 2, v * (float)hv[2]);
    atomicAdd(op + 3, v * (float)hv[3]);
}

// ---------------------------------------------------------------------------
extern "C" void kernel_launch(void* const* d_in, const int* in_sizes, int n_in,
                              void* d_out, int out_size, void* d_ws, size_t ws_size,
                              hipStream_t stream) {
    const float* x         = (const float*)d_in[0];   // [1,100000,256]
    const float* W         = (const float*)d_in[1];   // [256,128]
    const float* edge_vals = (const float*)d_in[2];   // [E]
    const int*   edge_rows = (const int*)d_in[3];     // [E]
    const int*   edge_cols = (const int*)d_in[4];     // [E]
    float*       out       = (float*)d_out;           // [100000,128] f32

    const int nEdges = in_sizes[2];
    const int nNodes = in_sizes[0] / K_DIM;           // 100000

    // Workspace: Wt f16 (64KB) | h f16 (nNodes*128*2 B ~ 25.6MB)
    _Float16* Wt = (_Float16*)d_ws;
    _Float16* h  = (_Float16*)((char*)d_ws + 65536);

    // Zero the accumulation target (harness poisons d_out).
    hipMemsetAsync(d_out, 0, (size_t)out_size * sizeof(float), stream);

    wt_convert_kernel<<<(K_DIM * N_DIM + 255) / 256, 256, 0, stream>>>(W, Wt);

    gemm_kernel<<<nNodes / M_TILE, 256, 0, stream>>>(x, Wt, h);

    long long totalT = (long long)nEdges * 32;
    int spmmBlocks = (int)((totalT + 255) / 256);
    spmm_kernel<<<spmmBlocks, 256, 0, stream>>>(edge_vals, edge_rows, edge_cols,
                                                h, out, nEdges);
}
